// MoEHead2_35622458753640
// MI455X (gfx1250) — matrix operand order repaired and checked by reading, not compiled, not run on hardware
//
#include <hip/hip_runtime.h>
#include <hip/hip_bf16.h>

typedef unsigned int u32;
typedef __bf16 bf16_t;
typedef __attribute__((ext_vector_type(16))) __bf16 v16bf;
typedef __attribute__((ext_vector_type(8)))  float  v8f;

#define DIM      1024
#define NEXP     8
#define TOKENS   2048          // B*S
#define SLOTS    4096          // TOKENS*K
#define HIDDEN   2730
#define UPOUT    5460
#define HPAD     2752          // HIDDEN padded to multiple of 64
#define VOCAB    16384
#define LDS_P    36            // LDS row pitch in uints (72 bf16)

union FragAB { v16bf v; u32 u[8]; };
union FragC  { v8f  v; float f[8]; };

__device__ __forceinline__ unsigned short f2bf(float f) {
    u32 u = __float_as_uint(f);
    u += 0x7fffu + ((u >> 16) & 1u);   // round-to-nearest-even
    return (unsigned short)(u >> 16);
}

__device__ __forceinline__ u32 pack2bf(float a, float b) {
    return (u32)f2bf(a) | ((u32)f2bf(b) << 16);
}

__device__ __forceinline__ v8f wmma_bf16(v16bf a, v16bf b, v8f c) {
    return __builtin_amdgcn_wmma_f32_16x16x32_bf16(
        false, a, false, b, (short)0, c, false, false);
}

__device__ __forceinline__ void async_wait0() {
    asm volatile("s_wait_asynccnt 0" ::: "memory");
}

// A fragment 16x32 bf16: lanes 0-15 -> M=lane (k 0..7 pattern), lanes 16-31 -> same M, k+8
__device__ __forceinline__ v16bf load_a_frag(const u32* As_u, int mbase, int ks, int lane) {
    FragAB fa;
    int m  = mbase + (lane & 15);
    int kh = (lane >> 4) * 8;
#pragma unroll
    for (int r = 0; r < 8; ++r) {
        int kk = ((r >= 4) ? 16 : 0) + kh + 2 * (r & 3);
        fa.u[r] = As_u[m * LDS_P + ((ks + kk) >> 1)];
    }
    return fa.v;
}

// B fragment 32x16 bf16: lanes 0-15 -> N=lane (k 0..7, 16..23), lanes 16-31 -> same N, k+8
__device__ __forceinline__ v16bf load_b_frag(const u32* Bs_u, int nbase, int ks, int lane) {
    FragAB fb;
    int n  = nbase + (lane & 15);
    int kh = (lane >> 4) * 8;
#pragma unroll
    for (int r = 0; r < 8; ++r) {
        int k = ((r >= 4) ? 16 : 0) + kh + 2 * (r & 3);
        fb.u[r] = Bs_u[n * LDS_P + ((ks + k) >> 1)];
    }
    return fb.v;
}

// stage 128 rows x 64 cols of bf16 A via async global->LDS copies (ASYNCcnt path).
// 16B chunks; rows past the slot buffer are zero-filled with direct LDS stores.
__device__ __forceinline__ void stage_a_async(u32* As_u, const unsigned short* A, size_t lda,
                                              int row0, int k0, int tid) {
    for (int i = tid; i < 128 * 8; i += 256) {          // 8 x 16B chunks per row
        int r = i >> 3, c = i & 7;
        int gr = row0 + r;
        u32* dst = &As_u[r * LDS_P + (c << 1)];
        if (gr < SLOTS) {
            const unsigned short* src = A + (size_t)gr * lda + k0 + (c << 3);
            u32 lds = (u32)(uintptr_t)dst;              // low 32b of flat ptr == LDS offset
            asm volatile("global_load_async_to_lds_b128 %0, %1, off"
                         :: "v"(lds), "v"(src) : "memory");
        } else {
            dst[0] = 0u; dst[1] = 0u; dst[2] = 0u; dst[3] = 0u;
        }
    }
}

// stage 128 n-rows x 64 k-cols of weight (fp32 [n][k], ldw mult of 4, no k-tail) -> bf16 LDS
__device__ __forceinline__ void stage_b_vec(u32* Bs_u, const float* W, int nrow0, int Nw,
                                            int k0, size_t ldw, int tid) {
    for (int i = tid; i < 128 * 16; i += 256) {         // 16 float4 per row
        int n = i >> 4, c4 = i & 15;
        int gn = nrow0 + n;
        u32 lo = 0u, hi = 0u;
        if (gn < Nw) {
            const float4 f = *(const float4*)(W + (size_t)gn * ldw + k0 + (c4 << 2));
            lo = pack2bf(f.x, f.y);
            hi = pack2bf(f.z, f.w);
        }
        Bs_u[n * LDS_P + (c4 << 1)]     = lo;
        Bs_u[n * LDS_P + (c4 << 1) + 1] = hi;
    }
}

// guarded scalar variant (unaligned ldw / k-tail), used by the down GEMM
__device__ __forceinline__ void stage_b(u32* Bs_u, const float* W, int nrow0, int Nw,
                                        int k0, int Kw, size_t ldw, int tid) {
    for (int i = tid; i < 128 * 32; i += 256) {
        int n = i >> 5, c = i & 31;
        int gn = nrow0 + n;
        u32 v = 0u;
        if (gn < Nw) {
            int gk = k0 + (c << 1);
            const float* p = W + (size_t)gn * ldw + gk;
            float f0 = (gk     < Kw) ? p[0] : 0.f;
            float f1 = (gk + 1 < Kw) ? p[1] : 0.f;
            v = pack2bf(f0, f1);
        }
        Bs_u[n * LDS_P + c] = v;
    }
}

// ---------------- router: scores, top-2, softmax weights, expert counts ----------------
__global__ __launch_bounds__(256) void router_kernel(const float* __restrict__ x,
                                                     const float* __restrict__ wr,
                                                     int* __restrict__ ids,
                                                     float* __restrict__ wts,
                                                     int* __restrict__ ctrl) {
    int t = blockIdx.x;
    int lane = threadIdx.x & 31, wv = threadIdx.x >> 5;
    const float* xr = x + (size_t)t * DIM;
    const float* we = wr + (size_t)wv * DIM;
    float s = 0.f;
    for (int d = lane; d < DIM; d += 32) s += xr[d] * we[d];
#pragma unroll
    for (int o = 16; o > 0; o >>= 1) s += __shfl_down(s, o, 32);
    __shared__ float sc[NEXP];
    if (lane == 0) sc[wv] = s;
    __syncthreads();
    if (threadIdx.x == 0) {
        int i0 = 0; float v0 = sc[0];
        for (int e = 1; e < NEXP; ++e) if (sc[e] > v0) { v0 = sc[e]; i0 = e; }
        int i1 = -1; float v1 = -3.4e38f;
        for (int e = 0; e < NEXP; ++e) if (e != i0 && sc[e] > v1) { v1 = sc[e]; i1 = e; }
        float e1 = __expf(v1 - v0);
        float w0 = 1.f / (1.f + e1);
        ids[2 * t] = i0; ids[2 * t + 1] = i1;
        wts[2 * t] = w0; wts[2 * t + 1] = 1.f - w0;
        atomicAdd(&ctrl[i0], 1);
        atomicAdd(&ctrl[i1], 1);
    }
}

// ---------------- exclusive prefix sum over 8 experts ----------------
__global__ void scan_kernel(int* ctrl) {
    if (threadIdx.x == 0) {
        int off = 0;
        for (int e = 0; e < NEXP; ++e) { ctrl[16 + e] = off; off += ctrl[e]; }
        ctrl[24] = off;
    }
}

// ---------------- scatter slots into grouped order + fp32->bf16 rows ----------------
__global__ __launch_bounds__(256) void scatter_kernel(const float* __restrict__ x,
                                                      const int* __restrict__ ids,
                                                      int* __restrict__ ctrl,
                                                      int* __restrict__ rwid,
                                                      unsigned short* __restrict__ xs) {
    int slot = blockIdx.x;
    __shared__ int s_pos;
    if (threadIdx.x == 0) {
        int e = ids[slot];
        int pos = ctrl[16 + e] + atomicAdd(&ctrl[8 + e], 1);
        rwid[pos] = slot;
        s_pos = pos;
    }
    __syncthreads();
    int pos = s_pos;
    int t = slot >> 1;
    const float* xr = x + (size_t)t * DIM;
    unsigned short* dst = xs + (size_t)pos * DIM;
    for (int i = threadIdx.x; i < DIM; i += 256) dst[i] = f2bf(xr[i]);
}

// ---------------- grouped up-proj GEMM + fused SwiGLU ----------------
__global__ __launch_bounds__(256) void up_swiglu_kernel(const unsigned short* __restrict__ xs,
                                                        const float* __restrict__ w_up,
                                                        unsigned short* __restrict__ hact,
                                                        const int* __restrict__ ctrl) {
    __shared__ u32 As_u[128 * LDS_P];
    __shared__ u32 Bl_u[128 * LDS_P];
    __shared__ u32 Bg_u[128 * LDS_P];
    int e = blockIdx.y, mt = blockIdx.z, nt = blockIdx.x;
    int cnt = ctrl[e];
    if (mt * 128 >= cnt) return;
    int row0  = ctrl[16 + e] + mt * 128;
    int ncol0 = nt * 128;
    const float* W = w_up + (size_t)e * UPOUT * DIM;
    int tid = threadIdx.x, lane = tid & 31, wv = tid >> 5;
    int wm = (wv & 3) * 32, wn = (wv >> 2) * 64;
    const v8f z = {0, 0, 0, 0, 0, 0, 0, 0};
    v8f accL[2][4], accG[2][4];
#pragma unroll
    for (int i = 0; i < 2; ++i)
#pragma unroll
        for (int j = 0; j < 4; ++j) { accL[i][j] = z; accG[i][j] = z; }

    for (int k0 = 0; k0 < DIM; k0 += 64) {
        __syncthreads();
        stage_a_async(As_u, xs, DIM, row0, k0, tid);
        stage_b_vec(Bl_u, W,                        ncol0, HIDDEN, k0, DIM, tid);
        stage_b_vec(Bg_u, W + (size_t)HIDDEN * DIM, ncol0, HIDDEN, k0, DIM, tid);
        async_wait0();
        __syncthreads();
#pragma unroll
        for (int ks = 0; ks < 64; ks += 32) {
            v16bf a0 = load_a_frag(As_u, wm,      ks, lane);
            v16bf a1 = load_a_frag(As_u, wm + 16, ks, lane);
#pragma unroll
            for (int j = 0; j < 4; ++j) {
                v16bf bl = load_b_frag(Bl_u, wn + j * 16, ks, lane);
                accL[0][j] = wmma_bf16(a0, bl, accL[0][j]);
                accL[1][j] = wmma_bf16(a1, bl, accL[1][j]);
                v16bf bg = load_b_frag(Bg_u, wn + j * 16, ks, lane);
                accG[0][j] = wmma_bf16(a0, bg, accG[0][j]);
                accG[1][j] = wmma_bf16(a1, bg, accG[1][j]);
            }
        }
    }
#pragma unroll
    for (int i = 0; i < 2; ++i)
#pragma unroll
        for (int j = 0; j < 4; ++j) {
            FragC cl, cg; cl.v = accL[i][j]; cg.v = accG[i][j];
#pragma unroll
            for (int r = 0; r < 8; ++r) {
                int rl  = wm + i * 16 + ((lane >> 4) << 3) + r;
                int col = ncol0 + wn + j * 16 + (lane & 15);
                if (mt * 128 + rl < cnt && col < HPAD) {
                    float v = 0.f;
                    if (col < HIDDEN) {
                        float lin = cl.f[r], g = cg.f[r];
                        v = lin * (g / (1.f + __expf(-g)));
                    }
                    hact[(size_t)(row0 + rl) * HPAD + col] = f2bf(v);
                }
            }
        }
}

// ---------------- grouped down GEMM + residual ----------------
__global__ __launch_bounds__(256) void down_kernel(const unsigned short* __restrict__ hact,
                                                   const float* __restrict__ w_down,
                                                   const float* __restrict__ x,
                                                   const int* __restrict__ rwid,
                                                   unsigned short* __restrict__ yact,
                                                   const int* __restrict__ ctrl) {
    __shared__ u32 As_u[128 * LDS_P];
    __shared__ u32 Bs_u[128 * LDS_P];
    int e = blockIdx.y, mt = blockIdx.z, nt = blockIdx.x;
    int cnt = ctrl[e];
    if (mt * 128 >= cnt) return;
    int row0  = ctrl[16 + e] + mt * 128;
    int ncol0 = nt * 128;
    const float* W = w_down + (size_t)e * DIM * HIDDEN;
    int tid = threadIdx.x, lane = tid & 31, wv = tid >> 5;
    int wm = (wv & 3) * 32, wn = (wv >> 2) * 64;
    const v8f z = {0, 0, 0, 0, 0, 0, 0, 0};
    v8f acc[2][4];
#pragma unroll
    for (int i = 0; i < 2; ++i)
#pragma unroll
        for (int j = 0; j < 4; ++j) acc[i][j] = z;

    for (int k0 = 0; k0 < HPAD; k0 += 64) {
        __syncthreads();
        stage_a_async(As_u, hact, HPAD, row0, k0, tid);
        stage_b(Bs_u, W, ncol0, DIM, k0, HIDDEN, HIDDEN, tid);
        async_wait0();
        __syncthreads();
#pragma unroll
        for (int ks = 0; ks < 64; ks += 32) {
            v16bf a0 = load_a_frag(As_u, wm,      ks, lane);
            v16bf a1 = load_a_frag(As_u, wm + 16, ks, lane);
#pragma unroll
            for (int j = 0; j < 4; ++j) {
                v16bf b = load_b_frag(Bs_u, wn + j * 16, ks, lane);
                acc[0][j] = wmma_bf16(a0, b, acc[0][j]);
                acc[1][j] = wmma_bf16(a1, b, acc[1][j]);
            }
        }
    }
#pragma unroll
    for (int i = 0; i < 2; ++i)
#pragma unroll
        for (int j = 0; j < 4; ++j) {
            FragC c; c.v = acc[i][j];
#pragma unroll
            for (int r = 0; r < 8; ++r) {
                int rl  = wm + i * 16 + ((lane >> 4) << 3) + r;
                int col = ncol0 + wn + j * 16 + (lane & 15);
                if (mt * 128 + rl < cnt) {
                    int gr = row0 + rl;
                    int t  = rwid[gr] >> 1;
                    float v = c.f[r] + x[(size_t)t * DIM + col];
                    yact[(size_t)gr * DIM + col] = f2bf(v);
                }
            }
        }
}

// ---------------- grouped vocab-proj GEMM + weighted combine ----------------
__global__ __launch_bounds__(256) void proj_kernel(const unsigned short* __restrict__ yact,
                                                   const float* __restrict__ w_proj,
                                                   const int* __restrict__ rwid,
                                                   const float* __restrict__ wts,
                                                   float* __restrict__ out,
                                                   const int* __restrict__ ctrl) {
    __shared__ u32 As_u[128 * LDS_P];
    __shared__ u32 Bs_u[128 * LDS_P];
    int e = blockIdx.y, mt = blockIdx.z, nt = blockIdx.x;
    int cnt = ctrl[e];
    if (mt * 128 >= cnt) return;
    int row0  = ctrl[16 + e] + mt * 128;
    int ncol0 = nt * 128;
    const float* W = w_proj + (size_t)e * VOCAB * DIM;
    int tid = threadIdx.x, lane = tid & 31, wv = tid >> 5;
    int wm = (wv & 3) * 32, wn = (wv >> 2) * 64;
    const v8f z = {0, 0, 0, 0, 0, 0, 0, 0};
    v8f acc[2][4];
#pragma unroll
    for (int i = 0; i < 2; ++i)
#pragma unroll
        for (int j = 0; j < 4; ++j) acc[i][j] = z;

    for (int k0 = 0; k0 < DIM; k0 += 64) {
        __syncthreads();
        stage_a_async(As_u, yact, DIM, row0, k0, tid);
        stage_b_vec(Bs_u, W, ncol0, VOCAB, k0, DIM, tid);
        async_wait0();
        __syncthreads();
#pragma unroll
        for (int ks = 0; ks < 64; ks += 32) {
            v16bf a0 = load_a_frag(As_u, wm,      ks, lane);
            v16bf a1 = load_a_frag(As_u, wm + 16, ks, lane);
#pragma unroll
            for (int j = 0; j < 4; ++j) {
                v16bf b = load_b_frag(Bs_u, wn + j * 16, ks, lane);
                acc[0][j] = wmma_bf16(a0, b, acc[0][j]);
                acc[1][j] = wmma_bf16(a1, b, acc[1][j]);
            }
        }
    }
#pragma unroll
    for (int i = 0; i < 2; ++i)
#pragma unroll
        for (int j = 0; j < 4; ++j) {
            FragC c; c.v = acc[i][j];
#pragma unroll
            for (int r = 0; r < 8; ++r) {
                int rl  = wm + i * 16 + ((lane >> 4) << 3) + r;
                int col = ncol0 + wn + j * 16 + (lane & 15);
                if (mt * 128 + rl < cnt) {
                    int gr  = row0 + rl;
                    int wid = rwid[gr];
                    int t   = wid >> 1;
                    atomicAdd(&out[(size_t)t * VOCAB + col], wts[wid] * c.f[r]);
                }
            }
        }
}

extern "C" void kernel_launch(void* const* d_in, const int* in_sizes, int n_in,
                              void* d_out, int out_size, void* d_ws, size_t ws_size,
                              hipStream_t stream) {
    const float* x      = (const float*)d_in[0];
    const float* wr     = (const float*)d_in[1];
    const float* w_up   = (const float*)d_in[2];
    const float* w_down = (const float*)d_in[3];
    const float* w_proj = (const float*)d_in[4];
    float* out = (float*)d_out;

    char* ws = (char*)d_ws;
    int*   ctrl = (int*)ws;                                   // counts[8], cursor[8], offsets[9]
    int*   ids  = (int*)(ws + 256);                           // 4096 ints
    float* wts  = (float*)(ws + 256 + 16384);                 // 4096 floats
    int*   rwid = (int*)(ws + 256 + 2 * 16384);               // 4096 ints
    unsigned short* xs   = (unsigned short*)(ws + 65536);                         // 8 MB
    unsigned short* hact = (unsigned short*)(ws + 65536 + 8388608);               // 22 MB
    unsigned short* yact = (unsigned short*)(ws + 65536 + 8388608 + 22544384);    // 8 MB

    hipMemsetAsync(ctrl, 0, 256, stream);
    hipMemsetAsync(out, 0, (size_t)TOKENS * VOCAB * sizeof(float), stream);

    router_kernel<<<TOKENS, 256, 0, stream>>>(x, wr, ids, wts, ctrl);
    scan_kernel<<<1, 32, 0, stream>>>(ctrl);
    scatter_kernel<<<SLOTS, 256, 0, stream>>>(x, ids, ctrl, rwid, xs);
    up_swiglu_kernel<<<dim3(22, NEXP, 32), 256, 0, stream>>>(xs, w_up, hact, ctrl);
    down_kernel<<<dim3(8, NEXP, 32), 256, 0, stream>>>(hact, w_down, x, rwid, yact, ctrl);
    proj_kernel<<<dim3(128, NEXP, 32), 256, 0, stream>>>(yact, w_proj, rwid, wts, out, ctrl);
}
